// MultiBlockSTABiLSTM_43473658970665
// MI455X (gfx1250) — matrix operand order repaired and checked by reading, not compiled, not run on hardware
//
#include <hip/hip_runtime.h>

// ---------------------------------------------------------------------------
// MultiBlockSTABiLSTM for MI455X (gfx1250).
// All dense matmuls run on v_wmma_f32_16x16x32_f16 (f16 in, f32 accumulate).
// Input index assumption: JAX pytree flatten order (dict keys sorted):
//   0: edge_index [2,E] i32
//   per block b (base B = 1+28*b):
//     B+0 final.b  B+1 final.ln_b  B+2 final.ln_g  B+3 final.w[128,128]
//     B+4 fuse.b   B+5 fuse.w[256,128]
//     B+6 a_dst    B+7 a_src  B+8 gat.b  B+9 gat.ob  B+10 gat.ow  B+11 gat.w
//     B+12 bk  B+13 bo  B+14 bq  B+15 bv
//     B+16 lstm_b.bhh B+17 .bih B+18 .whh[256,64] B+19 .wih[256,128]
//     B+20 lstm_f.bhh B+21 .bih B+22 .whh        B+23 .wih
//     B+24 wk  B+25 wo  B+26 wq  B+27 wv
//   57 in.b  58 in.w[64,128]  59 out.b  60 out.ln_b  61 out.ln_g  62 out.w[128,64]
//   63 x [24,2000,64] f32
// ---------------------------------------------------------------------------

typedef __attribute__((ext_vector_type(16))) _Float16 v16h;
typedef __attribute__((ext_vector_type(8)))  _Float16 v8h;
typedef __attribute__((ext_vector_type(8)))  float    v8f;

static constexpr int TT    = 24;      // timesteps
static constexpr int NODES = 2000;    // graph nodes
static constexpr int FIN   = 64;      // input/output feature dim
static constexpr int HD    = 128;     // hidden dim
static constexpr int TN    = TT * NODES;

// ---------------------------------------------------------------------------
// Generic WMMA GEMM:  out[M,Ncols] = act( A[M,K](f16) * Wt[Ncols,K]^T + bias + addend )
// 8 waves per 256-thread block; each wave owns one 16x16 output tile.
// ---------------------------------------------------------------------------
__global__ __launch_bounds__(256)
void k_gemm(const _Float16* __restrict__ A, int lda,
            const _Float16* __restrict__ W, int K,
            const float* __restrict__ bias,
            const float* __restrict__ addend, int ldadd,
            float* __restrict__ outF, int ldf,
            _Float16* __restrict__ outH, int ldh,
            int M, int act)
{
  const int wave = threadIdx.x >> 5;
  const int lane = threadIdx.x & 31;
  const int mt   = blockIdx.x * 2 + (wave & 1);
  const int row0 = mt << 4;
  if (row0 >= M) return;                         // wave-uniform: EXEC stays all-1
  const int nt   = blockIdx.y * 4 + (wave >> 1);
  const int col0 = nt << 4;
  const int l15  = lane & 15;
  const int hsel = lane >> 4;

  // A fragment: lanes0-15 rows, K {0..15};
  // lanes16-31 same rows, K {16..31} (matches B's contiguous k order).
  const _Float16* ap = A + (size_t)(row0 + l15) * lda + hsel * 16;
  // B fragment: lane n<16 holds B[k..k+15][n]; lanes 16-31 hold B[k+16..k+31][n].
  // Wt is [Ncols][K] so this is one contiguous 32B load.
  const _Float16* bp = W + (size_t)(col0 + l15) * K + hsel * 16;

  v8f c = {};
  for (int k = 0; k < K; k += 32) {
    v16h b = *(const v16h*)(bp + k);
    v16h a = *(const v16h*)(ap + k);
    c = __builtin_amdgcn_wmma_f32_16x16x32_f16(false, a, false, b,
                                               (short)0, c, false, false);
  }

  const int col = col0 + l15;
  const float bc = bias ? bias[col] : 0.f;
#pragma unroll
  for (int r = 0; r < 8; ++r) {
    const int row = row0 + r + hsel * 8;         // C layout: VGPR r -> M=r / M=8+r
    float v = c[r] + bc;
    if (addend) v += addend[(size_t)row * ldadd + col];
    if (act)    v = v > 0.f ? v : 0.f;
    if (outF)   outF[(size_t)row * ldf + col] = v;
    if (outH)   outH[(size_t)row * ldh + col] = (_Float16)v;
  }
}

// ---------------------------------------------------------------------------
// Elementwise helpers
// ---------------------------------------------------------------------------
__global__ void k_cvt(const float* __restrict__ in, _Float16* __restrict__ out,
                      size_t n, int act)   // act: 0 none, 1 relu, 2 elu
{
  size_t i = (size_t)blockIdx.x * blockDim.x + threadIdx.x;
  if (i >= n) return;
  float v = in[i];
  if (act == 1)      v = fmaxf(v, 0.f);
  else if (act == 2) v = v > 0.f ? v : (__expf(v) - 1.f);
  out[i] = (_Float16)v;
}

__global__ void k_transpose_cvt(const float* __restrict__ in,   // [K][Nn]
                                _Float16* __restrict__ out,     // [Nn][K]
                                int K, int Nn)
{
  int i = blockIdx.x * blockDim.x + threadIdx.x;
  if (i >= K * Nn) return;
  int n = i % Nn, k = i / Nn;
  out[(size_t)n * K + k] = (_Float16)in[i];
}

__global__ void k_copy(const float* __restrict__ a, float* __restrict__ o, int n)
{
  int i = blockIdx.x * blockDim.x + threadIdx.x;
  if (i < n) o[i] = a[i];
}

__global__ void k_addv(const float* __restrict__ a, const float* __restrict__ b,
                       float* __restrict__ o, int n)
{
  int i = blockIdx.x * blockDim.x + threadIdx.x;
  if (i < n) o[i] = a[i] + b[i];
}

__global__ void k_fill(float* __restrict__ p, float v, size_t n)
{
  size_t i = (size_t)blockIdx.x * blockDim.x + threadIdx.x;
  if (i < n) p[i] = v;
}

__global__ void k_fillh(_Float16* __restrict__ p, size_t n)
{
  size_t i = (size_t)blockIdx.x * blockDim.x + threadIdx.x;
  if (i < n) p[i] = (_Float16)0.f;
}

// ---------------------------------------------------------------------------
// GAT kernels
// ---------------------------------------------------------------------------
__global__ void k_gat_scores(const float* __restrict__ wh,     // [TN][128]
                             const float* __restrict__ a_src,  // [4][32]
                             const float* __restrict__ a_dst,
                             float* __restrict__ es, float* __restrict__ ed,
                             int tn)
{
  int i = blockIdx.x * blockDim.x + threadIdx.x;   // (t*N+n)*4 + h
  if (i >= tn * 4) return;
  int h = i & 3, r = i >> 2;
  const float* w  = wh + (size_t)r * HD + h * 32;
  const float* as = a_src + h * 32;
  const float* ad = a_dst + h * 32;
  float s = 0.f, d = 0.f;
#pragma unroll
  for (int k = 0; k < 32; ++k) { float v = w[k]; s += v * as[k]; d += v * ad[k]; }
  es[i] = s; ed[i] = d;
}

__device__ inline void atomicMaxF(float* addr, float val)
{
  if (val >= 0.f) atomicMax((int*)addr, __float_as_int(val));
  else            atomicMin((unsigned int*)addr, __float_as_uint(val));
}

__global__ void k_edge_logits(const int* __restrict__ src, const int* __restrict__ dst,
                              const float* __restrict__ es, const float* __restrict__ ed,
                              float* __restrict__ ee, float* __restrict__ m, int E)
{
  int i = blockIdx.x * blockDim.x + threadIdx.x;   // e*T + t
  if (i >= E * TT) return;
  int t = i % TT, e = i / TT;
  int s = src[e], d = dst[e];
#pragma unroll
  for (int h = 0; h < 4; ++h) {
    float v = es[((size_t)t * NODES + s) * 4 + h] + ed[((size_t)t * NODES + d) * 4 + h];
    v = v > 0.f ? v : 0.2f * v;                    // leaky_relu(0.2)
    ee[(size_t)i * 4 + h] = v;
    atomicMaxF(&m[((size_t)d * TT + t) * 4 + h], v);
  }
}

__global__ void k_edge_exp(const int* __restrict__ dst, float* __restrict__ ee,
                           const float* __restrict__ m, float* __restrict__ z, int E)
{
  int i = blockIdx.x * blockDim.x + threadIdx.x;
  if (i >= E * TT) return;
  int t = i % TT, e = i / TT;
  int d = dst[e];
#pragma unroll
  for (int h = 0; h < 4; ++h) {
    float ex = __expf(ee[(size_t)i * 4 + h] - m[((size_t)d * TT + t) * 4 + h]);
    ee[(size_t)i * 4 + h] = ex;
    atomicAdd(&z[((size_t)d * TT + t) * 4 + h], ex);
  }
}

__global__ void k_edge_agg(const int* __restrict__ src, const int* __restrict__ dst,
                           const float* __restrict__ ee, const float* __restrict__ z,
                           const float* __restrict__ wh, float* __restrict__ agg, int E)
{
  int i = blockIdx.x * blockDim.x + threadIdx.x;   // (e*T + t)*4 + h
  if (i >= E * TT * 4) return;
  int h = i & 3, et = i >> 2;
  int t = et % TT, e = et / TT;
  int s = src[e], d = dst[e];
  float alpha = ee[i] / (z[((size_t)d * TT + t) * 4 + h] + 1e-16f);
  const float* w = wh + ((size_t)t * NODES + s) * HD + h * 32;
  float*       a = agg + ((size_t)t * NODES + d) * HD + h * 32;
#pragma unroll
  for (int k = 0; k < 32; ++k) atomicAdd(&a[k], alpha * w[k]);
}

// ---------------------------------------------------------------------------
// LSTM cell (after gates GEMM).  PyTorch order i,f,g,o.
// ---------------------------------------------------------------------------
__global__ void k_lstm_cell(const float* __restrict__ gates,  // [N][256]
                            float* __restrict__ c,            // [N][64]
                            _Float16* __restrict__ hst,       // [N][64]
                            _Float16* __restrict__ hcat,      // [T][N][128]
                            int tout, int diroff)
{
  int i = blockIdx.x * blockDim.x + threadIdx.x;
  if (i >= NODES * 64) return;
  int n = i >> 6, j = i & 63;
  const float* g = gates + (size_t)n * 256;
  float ig = g[j], fg = g[64 + j], gg = g[128 + j], og = g[192 + j];
  float si = 1.f / (1.f + __expf(-ig));
  float sf = 1.f / (1.f + __expf(-fg));
  float so = 1.f / (1.f + __expf(-og));
  float cc = sf * c[i] + si * tanhf(gg);
  float hh = so * tanhf(cc);
  c[i] = cc;
  hst[i] = (_Float16)hh;
  hcat[((size_t)tout * NODES + n) * HD + diroff + j] = (_Float16)hh;
}

// ---------------------------------------------------------------------------
// Per-node temporal attention: heads=4, dh=32, T=24. One block per node,
// threads 0..95 map to (head, t).
// ---------------------------------------------------------------------------
__global__ __launch_bounds__(128)
void k_attn(const _Float16* __restrict__ qkv,   // [T*N][384] (q|k|v)
            _Float16* __restrict__ out)         // [T*N][128]
{
  int n = blockIdx.x;
  int tid = threadIdx.x;
  if (tid >= 96) return;
  int hd = tid / TT, t = tid % TT;
  const float scale = 0.1767766952966369f;      // 1/sqrt(32)

  float q[32];
  const _Float16* qp = qkv + ((size_t)t * NODES + n) * 384 + hd * 32;
#pragma unroll
  for (int d = 0; d < 32; ++d) q[d] = (float)qp[d];

  float s[TT];
  float mx = -3.0e38f;
#pragma unroll
  for (int u = 0; u < TT; ++u) {
    const _Float16* kp = qkv + ((size_t)u * NODES + n) * 384 + 128 + hd * 32;
    float acc = 0.f;
#pragma unroll
    for (int d = 0; d < 32; ++d) acc += q[d] * (float)kp[d];
    acc *= scale;
    s[u] = acc;
    mx = fmaxf(mx, acc);
  }
  float zz = 0.f;
#pragma unroll
  for (int u = 0; u < TT; ++u) { s[u] = __expf(s[u] - mx); zz += s[u]; }
  float inv = 1.f / zz;

  float o[32];
#pragma unroll
  for (int d = 0; d < 32; ++d) o[d] = 0.f;
#pragma unroll
  for (int u = 0; u < TT; ++u) {
    const _Float16* vp = qkv + ((size_t)u * NODES + n) * 384 + 256 + hd * 32;
    float a = s[u] * inv;
#pragma unroll
    for (int d = 0; d < 32; ++d) o[d] += a * (float)vp[d];
  }
  _Float16* op = out + ((size_t)t * NODES + n) * HD + hd * 32;
#pragma unroll
  for (int d = 0; d < 32; ++d) op[d] = (_Float16)o[d];
}

// ---------------------------------------------------------------------------
// LayerNorm over last dim D (32|64|128), one wave per row.
// ---------------------------------------------------------------------------
__global__ __launch_bounds__(256)
void k_layernorm(const float* __restrict__ in, const float* __restrict__ g,
                 const float* __restrict__ bvec, float* __restrict__ outF,
                 _Float16* __restrict__ outH, int D, int M)
{
  int wave = threadIdx.x >> 5, lane = threadIdx.x & 31;
  int row = blockIdx.x * 8 + wave;
  if (row >= M) return;
  const float* xr = in + (size_t)row * D;
  int per = D >> 5;
  float v[4];
  float s = 0.f;
  for (int i = 0; i < per; ++i) { v[i] = xr[lane + (i << 5)]; s += v[i]; }
  for (int o = 16; o > 0; o >>= 1) s += __shfl_xor(s, o, 32);
  float mean = s / (float)D;
  float qq = 0.f;
  for (int i = 0; i < per; ++i) { float d = v[i] - mean; qq += d * d; }
  for (int o = 16; o > 0; o >>= 1) qq += __shfl_xor(qq, o, 32);
  float r = rsqrtf(qq / (float)D + 1e-5f);
  for (int i = 0; i < per; ++i) {
    int c = lane + (i << 5);
    float y = (v[i] - mean) * r * g[c] + bvec[c];
    if (outF) outF[(size_t)row * D + c] = y;
    if (outH) outH[(size_t)row * D + c] = (_Float16)y;
  }
}

// ---------------------------------------------------------------------------
// Host orchestration
// ---------------------------------------------------------------------------
static inline dim3 g1(size_t n) { return dim3((unsigned)((n + 255) / 256)); }

static inline void gemm(hipStream_t st, const _Float16* A, int lda,
                        const _Float16* W, int K, int Ncols,
                        const float* bias, const float* addend, int ldadd,
                        float* outF, int ldf, _Float16* outH, int ldh,
                        int M, int act)
{
  dim3 grid((unsigned)((M / 16 + 1) / 2), (unsigned)(Ncols / 64));
  k_gemm<<<grid, 256, 0, st>>>(A, lda, W, K, bias, addend, ldadd,
                               outF, ldf, outH, ldh, M, act);
}

extern "C" void kernel_launch(void* const* d_in, const int* in_sizes, int n_in,
                              void* d_out, int out_size, void* d_ws, size_t ws_size,
                              hipStream_t stream)
{
  (void)n_in; (void)out_size; (void)ws_size;
  const int*   edge = (const int*)d_in[0];
  const int    E    = in_sizes[0] / 2;
  const int*   esrc = edge;
  const int*   edst = edge + E;
  const float* x    = (const float*)d_in[63];
  auto P = [&](int i) { return (const float*)d_in[i]; };

  char*  base = (char*)d_ws;
  size_t off  = 0;
  auto alloc = [&](size_t bytes) -> char* {
    off = (off + 255) & ~(size_t)255;
    char* p = base + off; off += bytes; return p;
  };

  // persistent buffers
  _Float16* h16b  = (_Float16*)alloc((size_t)TN * HD * 2);       // block input (f16)
  _Float16* cat16 = (_Float16*)alloc((size_t)TN * 2 * HD * 2);   // [sp|tm] concat

  // converted weights
  _Float16 *gatWt[2], *owWt[2], *qkvWt[2], *woWt[2], *fuseWt[2], *finWt[2];
  _Float16 *wihF[2], *whhF[2], *wihB[2], *whhB[2];
  float *qkvB[2], *lbF[2], *lbB[2];
  for (int b = 0; b < 2; ++b) {
    gatWt[b]  = (_Float16*)alloc((size_t)HD * HD * 2);
    owWt[b]   = (_Float16*)alloc((size_t)HD * HD * 2);
    qkvWt[b]  = (_Float16*)alloc((size_t)3 * HD * HD * 2);
    woWt[b]   = (_Float16*)alloc((size_t)HD * HD * 2);
    fuseWt[b] = (_Float16*)alloc((size_t)2 * HD * HD * 2);
    finWt[b]  = (_Float16*)alloc((size_t)HD * HD * 2);
    wihF[b]   = (_Float16*)alloc((size_t)256 * HD * 2);
    whhF[b]   = (_Float16*)alloc((size_t)256 * 64 * 2);
    wihB[b]   = (_Float16*)alloc((size_t)256 * HD * 2);
    whhB[b]   = (_Float16*)alloc((size_t)256 * 64 * 2);
    qkvB[b]   = (float*)alloc(384 * 4);
    lbF[b]    = (float*)alloc(256 * 4);
    lbB[b]    = (float*)alloc(256 * 4);
  }
  _Float16* inWt  = (_Float16*)alloc((size_t)HD * FIN * 2);
  _Float16* outWt = (_Float16*)alloc((size_t)FIN * HD * 2);

  off = (off + 255) & ~(size_t)255;
  char* arena = base + off;   // phase-reused scratch

  // ---- weight conversion (every call; deterministic) ----
  for (int b = 0; b < 2; ++b) {
    int B = 1 + b * 28;
    k_transpose_cvt<<<g1(HD * HD), 256, 0, stream>>>(P(B + 11), gatWt[b], HD, HD);
    k_transpose_cvt<<<g1(HD * HD), 256, 0, stream>>>(P(B + 10), owWt[b], HD, HD);
    k_transpose_cvt<<<g1(HD * HD), 256, 0, stream>>>(P(B + 26), qkvWt[b],             HD, HD); // wq
    k_transpose_cvt<<<g1(HD * HD), 256, 0, stream>>>(P(B + 24), qkvWt[b] + HD * HD,   HD, HD); // wk
    k_transpose_cvt<<<g1(HD * HD), 256, 0, stream>>>(P(B + 27), qkvWt[b] + 2 * HD * HD, HD, HD); // wv
    k_transpose_cvt<<<g1(HD * HD), 256, 0, stream>>>(P(B + 25), woWt[b], HD, HD);
    k_transpose_cvt<<<g1(2 * HD * HD), 256, 0, stream>>>(P(B + 5), fuseWt[b], 2 * HD, HD);
    k_transpose_cvt<<<g1(HD * HD), 256, 0, stream>>>(P(B + 3), finWt[b], HD, HD);
    k_cvt<<<g1((size_t)256 * HD), 256, 0, stream>>>(P(B + 23), wihF[b], (size_t)256 * HD, 0);
    k_cvt<<<g1((size_t)256 * 64), 256, 0, stream>>>(P(B + 22), whhF[b], (size_t)256 * 64, 0);
    k_cvt<<<g1((size_t)256 * HD), 256, 0, stream>>>(P(B + 19), wihB[b], (size_t)256 * HD, 0);
    k_cvt<<<g1((size_t)256 * 64), 256, 0, stream>>>(P(B + 18), whhB[b], (size_t)256 * 64, 0);
    k_copy<<<g1(HD), 256, 0, stream>>>(P(B + 14), qkvB[b],            HD);   // bq
    k_copy<<<g1(HD), 256, 0, stream>>>(P(B + 12), qkvB[b] + HD,       HD);   // bk
    k_copy<<<g1(HD), 256, 0, stream>>>(P(B + 15), qkvB[b] + 2 * HD,   HD);   // bv
    k_addv<<<g1(256), 256, 0, stream>>>(P(B + 21), P(B + 20), lbF[b], 256);  // bih+bhh fwd
    k_addv<<<g1(256), 256, 0, stream>>>(P(B + 17), P(B + 16), lbB[b], 256);  // bih+bhh bwd
  }
  k_transpose_cvt<<<g1(FIN * HD), 256, 0, stream>>>(P(58), inWt, FIN, HD);
  k_transpose_cvt<<<g1(FIN * HD), 256, 0, stream>>>(P(62), outWt, HD, FIN);

  // ---- input projection: h = relu(x @ in_w + in_b) ----
  _Float16* x16 = (_Float16*)arena;
  k_cvt<<<g1((size_t)TN * FIN), 256, 0, stream>>>(x, x16, (size_t)TN * FIN, 0);
  gemm(stream, x16, FIN, inWt, FIN, HD, P(57), nullptr, 0,
       nullptr, 0, h16b, HD, TN, 1);

  for (int b = 0; b < 2; ++b) {
    int B = 1 + b * 28;

    // ---------------- GAT ----------------
    char* a = arena;
    float*    wh32  = (float*)a;     a += (size_t)TN * HD * 4;
    float*    es    = (float*)a;     a += (size_t)TN * 4 * 4;
    float*    ed    = (float*)a;     a += (size_t)TN * 4 * 4;
    float*    ee    = (float*)a;     a += (size_t)E * TT * 4 * 4;
    float*    mbuf  = (float*)a;     a += (size_t)NODES * TT * 4 * 4;
    float*    zbuf  = (float*)a;     a += (size_t)NODES * TT * 4 * 4;
    float*    agg   = (float*)a;     a += (size_t)TN * HD * 4;
    _Float16* agg16 = (_Float16*)a;  a += (size_t)TN * HD * 2;

    gemm(stream, h16b, HD, gatWt[b], HD, HD, P(B + 8), nullptr, 0,
         wh32, HD, nullptr, 0, TN, 0);
    k_gat_scores<<<g1((size_t)TN * 4), 256, 0, stream>>>(wh32, P(B + 7), P(B + 6), es, ed, TN);
    k_fill<<<g1((size_t)NODES * TT * 4), 256, 0, stream>>>(mbuf, -3.0e38f, (size_t)NODES * TT * 4);
    k_fill<<<g1((size_t)NODES * TT * 4), 256, 0, stream>>>(zbuf, 0.f, (size_t)NODES * TT * 4);
    k_fill<<<g1((size_t)TN * HD), 256, 0, stream>>>(agg, 0.f, (size_t)TN * HD);
    k_edge_logits<<<g1((size_t)E * TT), 256, 0, stream>>>(esrc, edst, es, ed, ee, mbuf, E);
    k_edge_exp<<<g1((size_t)E * TT), 256, 0, stream>>>(edst, ee, mbuf, zbuf, E);
    k_edge_agg<<<g1((size_t)E * TT * 4), 256, 0, stream>>>(esrc, edst, ee, zbuf, wh32, agg, E);
    k_cvt<<<g1((size_t)TN * HD), 256, 0, stream>>>(agg, agg16, (size_t)TN * HD, 2);  // elu
    gemm(stream, agg16, HD, owWt[b], HD, HD, P(B + 9), nullptr, 0,
         nullptr, 0, cat16, 2 * HD, TN, 0);        // sp -> concat cols [0,128)

    // ---------------- BiLSTM + temporal attention (arena reused) ----------------
    a = arena;
    float*    Xg    = (float*)a;     a += (size_t)TN * 256 * 4;
    float*    gates = (float*)a;     a += (size_t)NODES * 256 * 4;
    float*    cst   = (float*)a;     a += (size_t)NODES * 64 * 4;
    _Float16* hst   = (_Float16*)a;  a += (size_t)NODES * 64 * 2;
    _Float16* hcat  = (_Float16*)a;  a += (size_t)TN * HD * 2;
    _Float16* qkv   = (_Float16*)a;  a += (size_t)TN * 384 * 2;
    _Float16* attn  = (_Float16*)a;  a += (size_t)TN * HD * 2;

    // forward direction
    gemm(stream, h16b, HD, wihF[b], HD, 256, lbF[b], nullptr, 0,
         Xg, 256, nullptr, 0, TN, 0);
    k_fill<<<g1((size_t)NODES * 64), 256, 0, stream>>>(cst, 0.f, (size_t)NODES * 64);
    k_fillh<<<g1((size_t)NODES * 64), 256, 0, stream>>>(hst, (size_t)NODES * 64);
    for (int t = 0; t < TT; ++t) {
      gemm(stream, hst, 64, whhF[b], 64, 256, nullptr, Xg + (size_t)t * NODES * 256, 256,
           gates, 256, nullptr, 0, NODES, 0);
      k_lstm_cell<<<g1((size_t)NODES * 64), 256, 0, stream>>>(gates, cst, hst, hcat, t, 0);
    }
    // backward direction
    gemm(stream, h16b, HD, wihB[b], HD, 256, lbB[b], nullptr, 0,
         Xg, 256, nullptr, 0, TN, 0);
    k_fill<<<g1((size_t)NODES * 64), 256, 0, stream>>>(cst, 0.f, (size_t)NODES * 64);
    k_fillh<<<g1((size_t)NODES * 64), 256, 0, stream>>>(hst, (size_t)NODES * 64);
    for (int s = 0; s < TT; ++s) {
      int t = TT - 1 - s;
      gemm(stream, hst, 64, whhB[b], 64, 256, nullptr, Xg + (size_t)t * NODES * 256, 256,
           gates, 256, nullptr, 0, NODES, 0);
      k_lstm_cell<<<g1((size_t)NODES * 64), 256, 0, stream>>>(gates, cst, hst, hcat, t, 64);
    }

    gemm(stream, hcat, HD, qkvWt[b], HD, 384, qkvB[b], nullptr, 0,
         nullptr, 0, qkv, 384, TN, 0);
    k_attn<<<dim3(NODES), 128, 0, stream>>>(qkv, attn);
    gemm(stream, attn, HD, woWt[b], HD, HD, P(B + 13), nullptr, 0,
         nullptr, 0, cat16 + HD, 2 * HD, TN, 0);   // tm -> concat cols [128,256)

    // ---------------- fuse / final / layernorm ----------------
    a = arena;
    _Float16* fuse16 = (_Float16*)a; a += (size_t)TN * HD * 2;
    float*    fin32  = (float*)a;    a += (size_t)TN * HD * 4;
    gemm(stream, cat16, 2 * HD, fuseWt[b], 2 * HD, HD, P(B + 4), nullptr, 0,
         nullptr, 0, fuse16, HD, TN, 1);
    gemm(stream, fuse16, HD, finWt[b], HD, HD, P(B + 0), nullptr, 0,
         fin32, HD, nullptr, 0, TN, 1);
    k_layernorm<<<dim3((TN + 7) / 8), 256, 0, stream>>>(fin32, P(B + 2), P(B + 1),
                                                        nullptr, h16b, HD, TN);
  }

  // ---- output projection + layernorm ----
  float* o32 = (float*)arena;
  gemm(stream, h16b, HD, outWt, HD, FIN, P(59), nullptr, 0,
       o32, FIN, nullptr, 0, TN, 0);
  k_layernorm<<<dim3((TN + 7) / 8), 256, 0, stream>>>(o32, P(61), P(60),
                                                      (float*)d_out, nullptr, FIN, TN);
}
